// MatchingNet_86741159510411
// MI455X (gfx1250) — compile-verified, matrix operand order checked by reading, not run on hardware
//
#include <hip/hip_runtime.h>

typedef __attribute__((ext_vector_type(16))) _Float16 v16h;
typedef __attribute__((ext_vector_type(8)))  float    v8f;

#define BATCH  4096
#define HID    2048

// ---------------------------------------------------------------------------
// Pack p,q (f32, [B,32,32] each) into X f16 [B,2048] interleaved:
// X[b, 2*e] = p[b,e], X[b, 2*e+1] = q[b,e]
// ---------------------------------------------------------------------------
__global__ __launch_bounds__(256)
void pack_x_kernel(const float* __restrict__ p, const float* __restrict__ q,
                   _Float16* __restrict__ X) {
  int idx = blockIdx.x * 256 + threadIdx.x;            // 0 .. BATCH*1024-1
  union { _Float16 h[2]; unsigned u; } pk;
  pk.h[0] = (_Float16)p[idx];
  pk.h[1] = (_Float16)q[idx];
  ((unsigned*)X)[idx] = pk.u;                          // halves 2*idx, 2*idx+1
}

// ---------------------------------------------------------------------------
// W (K x N, f32, row-major) -> Wt (N x K, f16, row-major)
// ---------------------------------------------------------------------------
__global__ __launch_bounds__(256)
void transpose_w_kernel(const float* __restrict__ W, _Float16* __restrict__ Wt,
                        int K, int N) {
  int t = blockIdx.x * 256 + threadIdx.x;              // 0 .. K*N-1
  int k = t / N;
  int n = t - k * N;
  Wt[(size_t)n * K + k] = (_Float16)W[t];
}

// ---------------------------------------------------------------------------
// Tiled WMMA GEMM:  out = act(A @ Wt^T + bias)
//   A  : M x K f16 row-major (M=4096, K=2048)
//   Wt : N x K f16 row-major
// Block = 256 threads (8 waves) -> 128x128 tile, K-step 64, LDS double-buffer.
// Waves arranged 4x2: wave owns a 32x64 sub-tile = 2x4 accumulators.
// Per 32-K sub-step: 2 A frags + 4 B frags (12 ds_b128) feed 8 WMMAs.
// LDS row stride padded to 72 halves (144 B) to kill bank conflicts.
// STORE_F32==0: LeakyReLU(0.01) + f16 store; ==1: raw f32 store.
// ---------------------------------------------------------------------------
template<int STORE_F32>
__global__ __launch_bounds__(256)
void gemm_kernel(const _Float16* __restrict__ A, const _Float16* __restrict__ Wt,
                 const float* __restrict__ bias, void* __restrict__ outp, int N) {
  const int K   = HID;
  const int LDT = 72;                                   // padded stride (halves)
  __shared__ __align__(16) _Float16 As[2][128 * LDT];
  __shared__ __align__(16) _Float16 Bs[2][128 * LDT];

  const int tid  = threadIdx.x;
  const int lane = tid & 31;
  const int wave = tid >> 5;
  const int wm   = wave >> 1;                           // 0..3 -> 32-row band
  const int wn   = wave & 1;                            // 0..1 -> 64-col band
  const int bm   = blockIdx.x * 128;
  const int bn   = blockIdx.y * 128;

  v8f acc[2][4] = {};

  const int fr = lane & 15;                             // frag row (M / N idx)
  const int kh = lane >> 4;                             // K-half select

  // fill assignment: tile = 128 rows x 64 halves = 1024 uint4; 4 per thread
  int frow[4], fcol[4];
#pragma unroll
  for (int i = 0; i < 4; ++i) {
    int v   = tid + i * 256;
    frow[i] = v >> 3;
    fcol[i] = (v & 7) * 8;
  }

  uint4 ra[4], rb[4];
#pragma unroll
  for (int i = 0; i < 4; ++i) {                         // prologue: tile kk=0
    ra[i] = *(const uint4*)(A  + (size_t)(bm + frow[i]) * K + fcol[i]);
    rb[i] = *(const uint4*)(Wt + (size_t)(bn + frow[i]) * K + fcol[i]);
  }
#pragma unroll
  for (int i = 0; i < 4; ++i) {
    *(uint4*)(&As[0][frow[i] * LDT + fcol[i]]) = ra[i];
    *(uint4*)(&Bs[0][frow[i] * LDT + fcol[i]]) = rb[i];
  }
  __syncthreads();

  const int S = K / 64;
  for (int s = 0; s < S; ++s) {
    const int cur = s & 1;
    if (s + 1 < S) {                                    // prefetch next tile
      const int kk = (s + 1) * 64;
#pragma unroll
      for (int i = 0; i < 4; ++i) {
        ra[i] = *(const uint4*)(A  + (size_t)(bm + frow[i]) * K + kk + fcol[i]);
        rb[i] = *(const uint4*)(Wt + (size_t)(bn + frow[i]) * K + kk + fcol[i]);
      }
    }
    // ---- compute on buffer `cur` ----
#pragma unroll
    for (int ks = 0; ks < 64; ks += 32) {
      union { v16h h; uint4 u[2]; } af[2], bf[4];
#pragma unroll
      for (int ti = 0; ti < 2; ++ti) {
        const _Float16* base =
            &As[cur][(wm * 32 + ti * 16 + fr) * LDT + ks + kh * 8];
        af[ti].u[0] = *(const uint4*)(base);
        af[ti].u[1] = *(const uint4*)(base + 16);
      }
#pragma unroll
      for (int tj = 0; tj < 4; ++tj) {
        const _Float16* base =
            &Bs[cur][(wn * 64 + tj * 16 + fr) * LDT + ks + kh * 8];
        bf[tj].u[0] = *(const uint4*)(base);
        bf[tj].u[1] = *(const uint4*)(base + 16);
      }
#pragma unroll
      for (int ti = 0; ti < 2; ++ti)
#pragma unroll
        for (int tj = 0; tj < 4; ++tj)
          acc[ti][tj] = __builtin_amdgcn_wmma_f32_16x16x32_f16(
              false, af[ti].h, false, bf[tj].h, (short)0, acc[ti][tj],
              false, false);
    }
    if (s + 1 < S) {                                    // stage into other buf
#pragma unroll
      for (int i = 0; i < 4; ++i) {
        *(uint4*)(&As[cur ^ 1][frow[i] * LDT + fcol[i]]) = ra[i];
        *(uint4*)(&Bs[cur ^ 1][frow[i] * LDT + fcol[i]]) = rb[i];
      }
    }
    __syncthreads();
  }

  // ---- epilogue: C/D layout -> VGPR v = M row v (lanes<16) / v+8 (lanes>=16),
  //      N = lane & 15 ----
#pragma unroll
  for (int ti = 0; ti < 2; ++ti)
#pragma unroll
    for (int tj = 0; tj < 4; ++tj)
#pragma unroll
      for (int v = 0; v < 8; ++v) {
        int mm   = v + ((lane >> 4) << 3);
        int nn   = lane & 15;
        int grow = bm + wm * 32 + ti * 16 + mm;
        int gcol = bn + wn * 64 + tj * 16 + nn;
        float val = acc[ti][tj][v] + bias[gcol];
        if (STORE_F32) {
          ((float*)outp)[(size_t)grow * N + gcol] = val;
        } else {
          val = val > 0.0f ? val : 0.01f * val;         // LeakyReLU
          ((_Float16*)outp)[(size_t)grow * N + gcol] = (_Float16)val;
        }
      }
}

// ---------------------------------------------------------------------------
// Softplus + 10 Sinkhorn iterations on each 32x32 matrix, in place in d_out.
// One 1024-thread block per batch element; matrix lives in LDS.
// ---------------------------------------------------------------------------
__global__ __launch_bounds__(1024)
void sinkhorn_kernel(float* __restrict__ R) {
  __shared__ float s[32][33];
  __shared__ float sums[32];
  const int b = blockIdx.x;
  const int i = threadIdx.x >> 5;
  const int j = threadIdx.x & 31;

  float x = R[(size_t)b * 1024 + i * 32 + j];
  // numerically-stable softplus = max(x,0) + log1p(exp(-|x|))
  s[i][j] = fmaxf(x, 0.0f) + log1pf(expf(-fabsf(x)));
  __syncthreads();

  for (int it = 0; it < 10; ++it) {
    if (threadIdx.x < 32) {                             // column sums (axis=1)
      float cs = 0.0f;
#pragma unroll
      for (int r = 0; r < 32; ++r) cs += fabsf(s[r][threadIdx.x]);
      sums[threadIdx.x] = fmaxf(cs, 1e-8f);
    }
    __syncthreads();
    s[i][j] = s[i][j] / sums[j];
    __syncthreads();
    if (threadIdx.x < 32) {                             // row sums (axis=2)
      float rs = 0.0f;
#pragma unroll
      for (int c = 0; c < 32; ++c) rs += fabsf(s[threadIdx.x][c]);
      sums[threadIdx.x] = fmaxf(rs, 1e-8f);
    }
    __syncthreads();
    s[i][j] = s[i][j] / sums[i];
    __syncthreads();
  }
  R[(size_t)b * 1024 + i * 32 + j] = s[i][j];
}

// ---------------------------------------------------------------------------
extern "C" void kernel_launch(void* const* d_in, const int* in_sizes, int n_in,
                              void* d_out, int out_size, void* d_ws, size_t ws_size,
                              hipStream_t stream) {
  (void)in_sizes; (void)n_in; (void)out_size; (void)ws_size;

  const float* p  = (const float*)d_in[0];
  const float* q  = (const float*)d_in[1];
  const float* W1 = (const float*)d_in[2];  const float* b1 = (const float*)d_in[3];
  const float* W2 = (const float*)d_in[4];  const float* b2 = (const float*)d_in[5];
  const float* W3 = (const float*)d_in[6];  const float* b3 = (const float*)d_in[7];
  const float* W4 = (const float*)d_in[8];  const float* b4 = (const float*)d_in[9];
  const float* W5 = (const float*)d_in[10]; const float* b5 = (const float*)d_in[11];

  char* ws = (char*)d_ws;
  const size_t ACT = (size_t)BATCH * HID * sizeof(_Float16);   // 16 MB
  const size_t WSQ = (size_t)HID * HID * sizeof(_Float16);     //  8 MB
  _Float16* X   = (_Float16*)ws;                 // ping buffer 0
  _Float16* H   = (_Float16*)(ws + ACT);         // ping buffer 1
  _Float16* Wt1 = (_Float16*)(ws + 2 * ACT);
  _Float16* Wt2 = (_Float16*)(ws + 2 * ACT + 1 * WSQ);
  _Float16* Wt3 = (_Float16*)(ws + 2 * ACT + 2 * WSQ);
  _Float16* Wt4 = (_Float16*)(ws + 2 * ACT + 3 * WSQ);
  _Float16* Wt5 = (_Float16*)(ws + 2 * ACT + 4 * WSQ);
  float* out = (float*)d_out;

  // input packing + weight transpose/convert
  pack_x_kernel<<<(BATCH * 1024) / 256, 256, 0, stream>>>(p, q, X);
  transpose_w_kernel<<<(HID * HID) / 256, 256, 0, stream>>>(W1, Wt1, HID, HID);
  transpose_w_kernel<<<(HID * HID) / 256, 256, 0, stream>>>(W2, Wt2, HID, HID);
  transpose_w_kernel<<<(HID * HID) / 256, 256, 0, stream>>>(W3, Wt3, HID, HID);
  transpose_w_kernel<<<(HID * HID) / 256, 256, 0, stream>>>(W4, Wt4, HID, HID);
  transpose_w_kernel<<<(HID * 1024) / 256, 256, 0, stream>>>(W5, Wt5, HID, 1024);

  dim3 blk(256);
  dim3 g1(BATCH / 128, HID / 128);    // 32 x 16
  dim3 g5(BATCH / 128, 1024 / 128);   // 32 x 8

  gemm_kernel<0><<<g1, blk, 0, stream>>>(X, Wt1, b1, H,   HID);
  gemm_kernel<0><<<g1, blk, 0, stream>>>(H, Wt2, b2, X,   HID);
  gemm_kernel<0><<<g1, blk, 0, stream>>>(X, Wt3, b3, H,   HID);
  gemm_kernel<0><<<g1, blk, 0, stream>>>(H, Wt4, b4, X,   HID);
  gemm_kernel<1><<<g5, blk, 0, stream>>>(X, Wt5, b5, out, 1024);

  sinkhorn_kernel<<<BATCH, 1024, 0, stream>>>(out);
}
